// ContextClusterModule_22273700397608
// MI455X (gfx1250) — matrix operand order repaired and checked by reading, not compiled, not run on hardware
//
#include <hip/hip_runtime.h>

// ---------------------------------------------------------------------------
// Types for CDNA5 WMMA (wave32, 16x16x32 bf16 -> f32 accumulate)
// ---------------------------------------------------------------------------
typedef __attribute__((ext_vector_type(16))) __bf16 v16bf;
typedef __attribute__((ext_vector_type(8)))  float  v8f;

#define HEADS 4
#define HEAD_DIM 24
#define M_PROTO 4
#define CIN 128
#define DFEAT 96        // HEADS*HEAD_DIM
#define DFV 192         // f(96) + v(96) fused output channels
#define COUT 128
#define NTOK 12544      // 112*112
#define WIMG 112
#define HIMG 112
#define NSLAB 64        // N tile per block (4 wmma column tiles)

__device__ __forceinline__ __bf16 f2bf(float f) {
    unsigned u = __builtin_bit_cast(unsigned, f);
    unsigned r = u + 0x7FFFu + ((u >> 16) & 1u);   // round-to-nearest-even
    unsigned short h = (unsigned short)(r >> 16);
    return __builtin_bit_cast(__bf16, h);
}
__device__ __forceinline__ float bf2f(__bf16 h) {
    unsigned short s = __builtin_bit_cast(unsigned short, h);
    unsigned u = ((unsigned)s) << 16;
    return __builtin_bit_cast(float, u);
}

// ---------------------------------------------------------------------------
// Kernel 0: weight conversion to bf16, center normalization, accumulator zero
// ---------------------------------------------------------------------------
__global__ __launch_bounds__(256) void prep_kernel(
    const float* __restrict__ f_w, const float* __restrict__ v_w,
    const float* __restrict__ proj_w, const float* __restrict__ centers,
    __bf16* __restrict__ wfv, __bf16* __restrict__ wproj, float* __restrict__ cn,
    float* __restrict__ aggsum, float* __restrict__ poolsum, float* __restrict__ simsum)
{
    const int tid = threadIdx.x;
    const int gstride = gridDim.x * blockDim.x;
    const int gid = blockIdx.x * blockDim.x + tid;

    // fused f/v weight: rows 0..95 = f_w, 96..191 = v_w  [192 x 128]
    for (int i = gid; i < DFV * CIN; i += gstride) {
        int o = i / CIN, k = i % CIN;
        float w = (o < DFEAT) ? f_w[o * CIN + k] : v_w[(o - DFEAT) * CIN + k];
        wfv[i] = f2bf(w);
    }
    // projection weight [128 x 96]
    for (int i = gid; i < COUT * DFEAT; i += gstride)
        wproj[i] = f2bf(proj_w[i]);
    // zero reduction accumulators (re-zeroed every launch: determinism)
    for (int i = gid; i < 64 * M_PROTO * HEAD_DIM; i += gstride) {
        aggsum[i] = 0.0f;
        poolsum[i] = 0.0f;
    }
    for (int i = gid; i < 64 * M_PROTO; i += gstride) simsum[i] = 0.0f;

    // normalized centers (block 0 only; uniform per-block branch)
    if (blockIdx.x == 0) {
        __shared__ float s_nrm[M_PROTO];
        if (tid < M_PROTO) {
            float s = 0.0f;
            for (int c = 0; c < HEAD_DIM; ++c) {
                float x = centers[tid * HEAD_DIM + c];
                s += x * x;
            }
            s_nrm[tid] = fmaxf(sqrtf(s), 1e-12f);
        }
        __syncthreads();
        if (tid < M_PROTO * HEAD_DIM)
            cn[tid] = centers[tid] / s_nrm[tid / HEAD_DIM];
    }
}

// ---------------------------------------------------------------------------
// Kernel 1: fused f/v 1x1-conv GEMM  fv[b, 192, n] = Wfv[192,128] @ x[b,128,n]
// 12 waves/block; each wave owns one 16-row M block; N slab = 64.
// ---------------------------------------------------------------------------
__global__ __launch_bounds__(384) void fv_gemm_kernel(
    const float* __restrict__ x, const float* __restrict__ f_b,
    const float* __restrict__ v_b, const __bf16* __restrict__ wfv,
    __bf16* __restrict__ fv, int n)
{
    __shared__ __bf16 Bs[CIN * NSLAB];   // x slab, pre-swizzled to B-frag layout
    const int b = blockIdx.y;
    const int hw0 = blockIdx.x * NSLAB;
    const int tid = threadIdx.x;

    // Stage x slab: convert f32 -> bf16, store in WMMA B-fragment order.
    // Fragment (ks, nt): 32 lanes * 16 contiguous bf16.
    // lane = (N & 15) + ((K & 16) ? 16 : 0), slot i = K & 15.
    const float* xb = x + (size_t)b * CIN * n + hw0;
    for (int e = tid; e < CIN * NSLAB; e += 384) {
        int k = e >> 6, col = e & 63;
        float val = xb[(size_t)k * n + col];
        int ks = k >> 5, kk = k & 31;
        int nt = col >> 4, cl = col & 15;
        int lane = cl + (kk & 16);
        Bs[(((ks << 2) | nt) * 32 + lane) * 16 + (kk & 15)] = f2bf(val);
    }
    __syncthreads();

    const int wave = tid >> 5;          // 0..11 -> M block
    const int lane = tid & 31;
    const int rowA = (wave << 4) + (lane & 15);
    const int kbaseA = (lane & 16) ? 16 : 0;  // k order used here: lane hi-half => K+16
    const int mhi = (lane & 16) ? 8 : 0;      // C/D layout: lane hi-half => M+8

    // Preload A fragments for all 4 K steps (contiguous k order k = 16*half + i, same as Bs).
    v16bf afrag[4];
    #pragma unroll
    for (int ks = 0; ks < 4; ++ks) {
        #pragma unroll
        for (int i = 0; i < 16; ++i) {
            int k = (ks << 5) + kbaseA + i;
            afrag[ks][i] = wfv[rowA * CIN + k];
        }
    }
    float biasv[8];
    #pragma unroll
    for (int r = 0; r < 8; ++r) {
        int Mrow = (wave << 4) + r + mhi;
        biasv[r] = (Mrow < DFEAT) ? f_b[Mrow] : v_b[Mrow - DFEAT];
    }

    #pragma unroll
    for (int nt = 0; nt < 4; ++nt) {
        v8f acc = {};
        #pragma unroll
        for (int ks = 0; ks < 4; ++ks) {
            const v16bf* bp = (const v16bf*)&Bs[(((ks << 2) | nt) * 32 + lane) * 16];
            acc = __builtin_amdgcn_wmma_f32_16x16x32_bf16(
                false, afrag[ks], false, *bp, (short)0, acc, false, false);
        }
        int col = hw0 + (nt << 4) + (lane & 15);
        #pragma unroll
        for (int r = 0; r < 8; ++r) {
            int Mrow = (wave << 4) + r + mhi;
            fv[((size_t)b * DFV + Mrow) * n + col] = f2bf(acc[r] + biasv[r]);
        }
    }
}

// ---------------------------------------------------------------------------
// Kernel 2: per-(b,head) assignment + global reduction. Computes cosine-sim /
// sigmoid / top-1 per token, packs (sim, idx) for reuse by the scatter pass,
// and accumulates sim sums, winner-weighted v aggregates, and 2x2 pool sums
// via LDS partials -> global f32 atomics.
// Pack format: f32 sim bits with the 2 low mantissa bits replaced by idx.
// ---------------------------------------------------------------------------
__global__ __launch_bounds__(256) void assign_reduce_kernel(
    const __bf16* __restrict__ fv, const float* __restrict__ cn,
    const float* __restrict__ alpha_p, const float* __restrict__ beta_p,
    float* __restrict__ aggsum, float* __restrict__ poolsum,
    float* __restrict__ simsum, unsigned* __restrict__ pack, int n)
{
    const int be = blockIdx.y, b = be >> 2, head = be & 3;
    const int tid = threadIdx.x;
    const int t = blockIdx.x * 256 + tid;   // 49*256 == 12544 exactly

    __shared__ float s_cn[M_PROTO * HEAD_DIM];
    __shared__ float s_agg[M_PROTO * HEAD_DIM];
    __shared__ float s_pool[M_PROTO * HEAD_DIM];
    __shared__ float s_sim[M_PROTO];
    if (tid < M_PROTO * HEAD_DIM) {
        s_cn[tid] = cn[tid];
        s_agg[tid] = 0.0f;
        s_pool[tid] = 0.0f;
    }
    if (tid < M_PROTO) s_sim[tid] = 0.0f;
    __syncthreads();

    const float alpha = alpha_p[0], beta = beta_p[0];
    const __bf16* fptr = fv + ((size_t)b * DFV + head * HEAD_DIM) * n + t;
    const __bf16* vptr = fv + ((size_t)b * DFV + DFEAT + head * HEAD_DIM) * n + t;

    // Put the v-planes in flight (global_prefetch_b8) while the f-side
    // norm / dot / sigmoid chain executes.
    #pragma unroll
    for (int c = 0; c < HEAD_DIM; ++c)
        __builtin_prefetch(vptr + (size_t)c * n, 0, 0);

    float fvec[HEAD_DIM];
    float nrm = 0.0f;
    #pragma unroll
    for (int c = 0; c < HEAD_DIM; ++c) {
        float fc = bf2f(fptr[(size_t)c * n]);
        fvec[c] = fc;
        nrm += fc * fc;
    }
    float inv = 1.0f / fmaxf(sqrtf(nrm), 1e-12f);
    int idx = 0;
    float simv = -1.0f;
    #pragma unroll
    for (int m = 0; m < M_PROTO; ++m) {
        float dot = 0.0f;
        #pragma unroll
        for (int c = 0; c < HEAD_DIM; ++c)
            dot += fvec[c] * s_cn[m * HEAD_DIM + c];
        float z = beta + alpha * dot * inv;
        float s = 1.0f / (1.0f + __expf(-z));
        if (s > simv) { simv = s; idx = m; }   // first-max like jnp.argmax
    }

    // Pack sim+idx for the scatter pass (drop 2 mantissa LSBs; rel err 2^-22).
    unsigned pbits = (__builtin_bit_cast(unsigned, simv) & 0xFFFFFFFCu) | (unsigned)idx;
    pack[(size_t)be * n + t] = pbits;
    const float simq = __builtin_bit_cast(float, pbits & 0xFFFFFFFCu);

    const int hrow = t / WIMG, wcol = t % WIMG;
    const int qm = ((hrow >= (HIMG / 2)) << 1) | (wcol >= (WIMG / 2));

    atomicAdd(&s_sim[idx], simq);
    #pragma unroll
    for (int c = 0; c < HEAD_DIM; ++c) {
        float vc = bf2f(vptr[(size_t)c * n]);
        atomicAdd(&s_agg[idx * HEAD_DIM + c], simq * vc);
        atomicAdd(&s_pool[qm * HEAD_DIM + c], vc);
    }
    __syncthreads();

    if (tid < M_PROTO * HEAD_DIM) {
        atomicAdd(&aggsum[be * M_PROTO * HEAD_DIM + tid], s_agg[tid]);
        atomicAdd(&poolsum[be * M_PROTO * HEAD_DIM + tid], s_pool[tid]);
    }
    if (tid < M_PROTO) atomicAdd(&simsum[be * M_PROTO + tid], s_sim[tid]);
}

// ---------------------------------------------------------------------------
// Kernel 3: finalize agg = (aggsum + pool_mean) / (simsum + 1)
// ---------------------------------------------------------------------------
__global__ __launch_bounds__(128) void finalize_kernel(
    const float* __restrict__ aggsum, const float* __restrict__ poolsum,
    const float* __restrict__ simsum, float* __restrict__ aggf)
{
    const int be = blockIdx.x, tid = threadIdx.x;
    if (tid < M_PROTO * HEAD_DIM) {
        const float invcnt = 1.0f / (float)((HIMG / 2) * (WIMG / 2));
        int m = tid / HEAD_DIM;
        aggf[be * M_PROTO * HEAD_DIM + tid] =
            (aggsum[be * M_PROTO * HEAD_DIM + tid] +
             poolsum[be * M_PROTO * HEAD_DIM + tid] * invcnt) /
            (simsum[be * M_PROTO + m] + 1.0f);
    }
}

// ---------------------------------------------------------------------------
// Kernel 4: scatter centers back to tokens using the packed (sim, idx):
// mid[b, head*24+c, t] = sim * agg[idx][c].  Pure streaming pass — no f reads,
// no sigmoid recompute.  Also emits the assignment map for be==0.
// ---------------------------------------------------------------------------
__global__ __launch_bounds__(256) void scatter_kernel(
    const unsigned* __restrict__ pack, const float* __restrict__ aggf,
    __bf16* __restrict__ mid, float* __restrict__ amap, int n)
{
    const int be = blockIdx.y, b = be >> 2, head = be & 3;
    const int tid = threadIdx.x;
    const int t = blockIdx.x * 256 + tid;

    __shared__ float s_ag[M_PROTO * HEAD_DIM];
    if (tid < M_PROTO * HEAD_DIM)
        s_ag[tid] = aggf[be * M_PROTO * HEAD_DIM + tid];
    __syncthreads();

    const unsigned pbits = pack[(size_t)be * n + t];
    const int idx = (int)(pbits & 3u);
    const float simv = __builtin_bit_cast(float, pbits & 0xFFFFFFFCu);

    __bf16* mptr = mid + ((size_t)b * DFEAT + head * HEAD_DIM) * n + t;
    #pragma unroll
    for (int c = 0; c < HEAD_DIM; ++c)
        mptr[(size_t)c * n] = f2bf(simv * s_ag[idx * HEAD_DIM + c]);

    if (be == 0) amap[t] = (float)idx;   // assignment_map = idx[0]
}

// ---------------------------------------------------------------------------
// Kernel 5: projection GEMM  out[b,128,n] = Wp[128,96] @ mid[b,96,n] + bias
// 8 waves/block, K = 96 (3 k-steps), N slab 64, f32 output.
// ---------------------------------------------------------------------------
__global__ __launch_bounds__(256) void proj_gemm_kernel(
    const __bf16* __restrict__ mid, const __bf16* __restrict__ wproj,
    const float* __restrict__ proj_b, float* __restrict__ out, int n)
{
    __shared__ __bf16 Bs[DFEAT * NSLAB];
    const int b = blockIdx.y;
    const int hw0 = blockIdx.x * NSLAB;
    const int tid = threadIdx.x;

    const __bf16* mb = mid + (size_t)b * DFEAT * n + hw0;
    for (int e = tid; e < DFEAT * NSLAB; e += 256) {
        int k = e >> 6, col = e & 63;
        __bf16 val = mb[(size_t)k * n + col];
        int ks = k >> 5, kk = k & 31;
        int nt = col >> 4, cl = col & 15;
        int lane = cl + (kk & 16);
        Bs[(((ks << 2) | nt) * 32 + lane) * 16 + (kk & 15)] = val;
    }
    __syncthreads();

    const int wave = tid >> 5;          // 0..7 -> M block (128 rows)
    const int lane = tid & 31;
    const int rowA = (wave << 4) + (lane & 15);
    const int kbaseA = (lane & 16) ? 16 : 0;
    const int mhi = (lane & 16) ? 8 : 0;

    v16bf afrag[3];
    #pragma unroll
    for (int ks = 0; ks < 3; ++ks) {
        #pragma unroll
        for (int i = 0; i < 16; ++i) {
            int k = (ks << 5) + kbaseA + i;
            afrag[ks][i] = wproj[rowA * DFEAT + k];
        }
    }

    #pragma unroll
    for (int nt = 0; nt < 4; ++nt) {
        v8f acc = {};
        #pragma unroll
        for (int ks = 0; ks < 3; ++ks) {
            const v16bf* bp = (const v16bf*)&Bs[(((ks << 2) | nt) * 32 + lane) * 16];
            acc = __builtin_amdgcn_wmma_f32_16x16x32_bf16(
                false, afrag[ks], false, *bp, (short)0, acc, false, false);
        }
        int col = hw0 + (nt << 4) + (lane & 15);
        #pragma unroll
        for (int r = 0; r < 8; ++r) {
            int Mrow = (wave << 4) + r + mhi;
            out[((size_t)b * COUT + Mrow) * n + col] = acc[r] + proj_b[Mrow];
        }
    }
}

// ---------------------------------------------------------------------------
// Host-side launcher
// ---------------------------------------------------------------------------
extern "C" void kernel_launch(void* const* d_in, const int* in_sizes, int n_in,
                              void* d_out, int out_size, void* d_ws, size_t ws_size,
                              hipStream_t stream) {
    (void)in_sizes; (void)n_in; (void)out_size; (void)ws_size;
    const float* x       = (const float*)d_in[0];
    const float* f_w     = (const float*)d_in[1];
    const float* f_b     = (const float*)d_in[2];
    const float* v_w     = (const float*)d_in[3];
    const float* v_b     = (const float*)d_in[4];
    const float* proj_w  = (const float*)d_in[5];
    const float* proj_b  = (const float*)d_in[6];
    const float* s_alpha = (const float*)d_in[7];
    const float* s_beta  = (const float*)d_in[8];
    const float* centers = (const float*)d_in[9];

    const int n = NTOK;
    const int B = 16;
    float* out = (float*)d_out;
    float* amap = out + (size_t)B * COUT * n;   // assignment map tail

    // Workspace layout (bytes, 256-aligned)
    char* ws = (char*)d_ws;
    __bf16*   wfv     = (__bf16*)(ws + 0);                    // 192*128*2 = 49152
    __bf16*   wproj   = (__bf16*)(ws + 49152);                // 128*96*2  = 24576
    float*    cn      = (float*)(ws + 73728);                 // 96*4      = 384
    float*    aggsum  = (float*)(ws + 74240);                 // 64*96*4   = 24576
    float*    poolsum = (float*)(ws + 98816);                 // 64*96*4   = 24576
    float*    simsum  = (float*)(ws + 123392);                // 64*4*4    = 1024
    float*    aggf    = (float*)(ws + 124416);                // 64*96*4   = 24576
    __bf16*   fvbuf   = (__bf16*)(ws + 148992);               // 16*192*n*2 = 77070336
    __bf16*   midbuf  = (__bf16*)(ws + 148992 + 77070336);    // 16*96*n*2  = 38535168
    unsigned* packbuf = (unsigned*)(ws + 148992 + 77070336 + 38535168); // 64*n*4 = 3211264

    // 0: prep weights / centers / zero accumulators
    prep_kernel<<<dim3(64), dim3(256), 0, stream>>>(
        f_w, v_w, proj_w, centers, wfv, wproj, cn, aggsum, poolsum, simsum);

    // 1: fused f/v GEMM (bf16 WMMA), 196 N-slabs x 16 batches
    fv_gemm_kernel<<<dim3(n / NSLAB, B), dim3(384), 0, stream>>>(
        x, f_b, v_b, wfv, fvbuf, n);

    // 2: cluster assignment + global reductions (49*256 == 12544 tokens)
    assign_reduce_kernel<<<dim3(n / 256, B * HEADS), dim3(256), 0, stream>>>(
        fvbuf, cn, s_alpha, s_beta, aggsum, poolsum, simsum, packbuf, n);

    // 3: finalize per-head center features
    finalize_kernel<<<dim3(B * HEADS), dim3(128), 0, stream>>>(
        aggsum, poolsum, simsum, aggf);

    // 4: redistribute centers to tokens (+ assignment map), reusing packed sim/idx
    scatter_kernel<<<dim3(n / 256, B * HEADS), dim3(256), 0, stream>>>(
        packbuf, aggf, midbuf, amap, n);

    // 5: projection GEMM (bf16 WMMA) straight into d_out
    proj_gemm_kernel<<<dim3(n / NSLAB, B), dim3(256), 0, stream>>>(
        midbuf, wproj, proj_b, out, n);
}
